// ScalarDotProductPolicyNetworkRevised_18614388261540
// MI455X (gfx1250) — matrix operand order repaired and checked by reading, not compiled, not run on hardware
//
#include <hip/hip_runtime.h>

// ---------------------------------------------------------------------------
// Fused scalar-dot-product policy network for MI455X (gfx1250, wave32, WMMA)
//   states[32768,8,128] -> attn1 -> attn2 -> MLP(256)+leakyrelu -> softmax(10)
// All weights live in LDS as bf16 (264KB, fits the 320KB/WGP LDS).
// Each wave processes a 16-row tile = 2 batches x 8 agents.
// ---------------------------------------------------------------------------

typedef __attribute__((ext_vector_type(16))) __bf16 v16bf;
typedef __attribute__((ext_vector_type(8)))  __bf16 v8bf;
typedef __attribute__((ext_vector_type(8)))  float  v8f;

union U16 { v16bf v; v8bf h[2]; };

#define TILES      16384            // (32768*8)/16 rows per tile
#define W_ELEMS    135168           // bf16 elems of weights in LDS
#define SCR_ELEMS  1536             // per-wave scratch (3 x 512)
#define SMEM_ELEMS (W_ELEMS + 8 * SCR_ELEMS)   // 147456 -> 294912 bytes

__device__ __forceinline__ v8f wmma_bf16(v16bf a, v16bf b, v8f c) {
  // D = A(16x32 bf16) * B(32x16 bf16) + C(16x16 f32)
  return __builtin_amdgcn_wmma_f32_16x16x32_bf16(false, a, false, b,
                                                 (short)0, c, false, false);
}

// A-fragment (16x32, interleaved K mapping): lane holds row (lane&15);
// elems 0..7 -> K = half*8 + i ; elems 8..15 -> K = 16 + half*8 + i
__device__ __forceinline__ v16bf ld_afrag(const __bf16* base, int stride,
                                          int rowbase, int kbase, int lane) {
  const int m = lane & 15, half = lane >> 4;
  const __bf16* p = base + (size_t)(rowbase + m) * stride + kbase + half * 8;
  U16 u;
  u.h[0] = *(const v8bf*)(p);
  u.h[1] = *(const v8bf*)(p + 16);
  return u.v;
}

// B-fragment (32x16, interleaved K mapping): lane holds col (lane&15);
// elems 0..7 -> K = half*8 + i ; elems 8..15 -> K = 16 + half*8 + i
__device__ __forceinline__ v16bf ld_bfrag(const __bf16* base, int stride,
                                          int colbase, int kbase, int lane) {
  const int n = lane & 15, half = lane >> 4;
  const __bf16* p = base + (size_t)(colbase + n) * stride + kbase + half * 8;
  U16 u;
  u.h[0] = *(const v8bf*)(p);
  u.h[1] = *(const v8bf*)(p + 16);
  return u.v;
}

// A-fragment straight from global f32 (states), converting to bf16.
__device__ __forceinline__ v16bf ld_afrag_global_f32(const float* __restrict__ base,
                                                     int rowbase, int kbase, int lane) {
  const int m = lane & 15, half = lane >> 4;
  const float* p = base + (size_t)(rowbase + m) * 128 + kbase + half * 8;
  const float4 a = *(const float4*)(p);
  const float4 b = *(const float4*)(p + 4);
  const float4 c = *(const float4*)(p + 16);
  const float4 d = *(const float4*)(p + 20);
  U16 u;
  u.v[0]  = (__bf16)a.x; u.v[1]  = (__bf16)a.y; u.v[2]  = (__bf16)a.z; u.v[3]  = (__bf16)a.w;
  u.v[4]  = (__bf16)b.x; u.v[5]  = (__bf16)b.y; u.v[6]  = (__bf16)b.z; u.v[7]  = (__bf16)b.w;
  u.v[8]  = (__bf16)c.x; u.v[9]  = (__bf16)c.y; u.v[10] = (__bf16)c.z; u.v[11] = (__bf16)c.w;
  u.v[12] = (__bf16)d.x; u.v[13] = (__bf16)d.y; u.v[14] = (__bf16)d.z; u.v[15] = (__bf16)d.w;
  return u.v;
}

// C-tile (16x16 f32, lane holds col n, rows vi+8*half) -> bf16 scratch [16][32]
__device__ __forceinline__ void store_ctile(__bf16* scr, v8f c, int cblock, int lane) {
  const int n = lane & 15, half = lane >> 4;
  __bf16* p = scr + (size_t)(8 * half) * 32 + cblock * 16 + n;
#pragma unroll
  for (int vi = 0; vi < 8; ++vi) p[vi * 32] = (__bf16)c[vi];
}

__device__ __forceinline__ float rowmax16(float v) {
#pragma unroll
  for (int m = 1; m < 16; m <<= 1) v = fmaxf(v, __shfl_xor(v, m, 16));
  return v;
}
__device__ __forceinline__ float rowsum16(float v) {
#pragma unroll
  for (int m = 1; m < 16; m <<= 1) v += __shfl_xor(v, m, 16);
  return v;
}

__device__ __forceinline__ void cp_w(__bf16* dst, const float* __restrict__ src,
                                     int n, int tid) {
  for (int i = tid * 4; i < n; i += 1024) {
    const float4 f = *(const float4*)(src + i);
    dst[i + 0] = (__bf16)f.x; dst[i + 1] = (__bf16)f.y;
    dst[i + 2] = (__bf16)f.z; dst[i + 3] = (__bf16)f.w;
  }
}

// One full attention layer for a 16-row tile held as 4 A-frags.
// Produces output as 4 A-frags; stores f32 softmax weights to w_out.
__device__ __forceinline__ void attention_layer(const v16bf* xa,
    const __bf16* Wq, const __bf16* Wk, const __bf16* Wv,
    float* __restrict__ w_out, int tile,
    __bf16* scrA, __bf16* scrB, __bf16* scrW,
    v16bf* oa, int lane)
{
  const int half = lane >> 4, col = lane & 15;
  v16bf vB[8];
  v8f s = {0, 0, 0, 0, 0, 0, 0, 0};

#pragma unroll
  for (int et = 0; et < 8; ++et) {
    v8f qc = {0, 0, 0, 0, 0, 0, 0, 0};
    v8f kc = {0, 0, 0, 0, 0, 0, 0, 0};
    v8f vc = {0, 0, 0, 0, 0, 0, 0, 0};
#pragma unroll
    for (int kf = 0; kf < 4; ++kf) {
      qc = wmma_bf16(xa[kf], ld_bfrag(Wq, 128, et * 16, kf * 32, lane), qc);
      kc = wmma_bf16(xa[kf], ld_bfrag(Wk, 128, et * 16, kf * 32, lane), kc);
      vc = wmma_bf16(xa[kf], ld_bfrag(Wv, 128, et * 16, kf * 32, lane), vc);
    }
    // v C-tile -> B-frag directly (rows j split by lane-half == padded-K split)
    U16 uv;
#pragma unroll
    for (int i = 0; i < 8; ++i) { uv.v[i] = (__bf16)vc[i]; uv.v[8 + i] = (__bf16)0.0f; }
    vB[et] = uv.v;
    // q,k need a lane transpose via LDS; accumulate scores per e-pair (K=32)
    store_ctile(scrA, qc, et & 1, lane);
    store_ctile(scrB, kc, et & 1, lane);
    if (et & 1) {
      const v16bf qa = ld_afrag(scrA, 32, 0, 0, lane);
      const v16bf kb = ld_bfrag(scrB, 32, 0, 0, lane);
      s = wmma_bf16(qa, kb, s);
    }
  }

  // Row-wise softmax over keys j; cross-batch blocks masked to -inf.
  const float scale = 0.08838834764831845f;       // 1/sqrt(128)
  const bool vl = ((col >> 3) == half);           // col's batch == row's batch
  float w[8];
#pragma unroll
  for (int vi = 0; vi < 8; ++vi) {
    const float v = vl ? s[vi] * scale : -1e30f;
    const float m = rowmax16(v);
    const float e = vl ? __expf(v - m) : 0.0f;
    const float sm = rowsum16(e);
    w[vi] = e / sm;
  }
  // Store f32 attention weights: w[(tile*2+half)*8 + i][j]
  if (vl) {
    const size_t base = (size_t)tile * 128 + (size_t)half * 64 + (col & 7);
#pragma unroll
    for (int vi = 0; vi < 8; ++vi) w_out[base + vi * 8] = w[vi];
  }

  // w -> A-frag (K j=0..7 in half0 elems 0..7, j=8..15 in half1 elems 0..7; elems 8..15 -> 0)
#pragma unroll
  for (int vi = 0; vi < 8; ++vi) scrW[(vi + 8 * half) * 16 + col] = (__bf16)w[vi];
  U16 uw;
  {
    const __bf16* p = scrW + (lane & 15) * 16;
    const v8bf lo = *(const v8bf*)(p);
    const v8bf hi = *(const v8bf*)(p + 8);
#pragma unroll
    for (int i = 0; i < 8; ++i) {
      uw.v[i]     = half ? hi[i] : lo[i];
      uw.v[8 + i] = (__bf16)0.0f;
    }
  }
  const v16bf wa = uw.v;

  // out = w @ v per e-tile; stream pairs through LDS into output A-frags
#pragma unroll
  for (int p = 0; p < 4; ++p) {
#pragma unroll
    for (int q = 0; q < 2; ++q) {
      v8f oc = {0, 0, 0, 0, 0, 0, 0, 0};
      oc = wmma_bf16(wa, vB[2 * p + q], oc);
      store_ctile(scrA, oc, q, lane);
    }
    oa[p] = ld_afrag(scrA, 32, 0, 0, lane);
  }
}

__global__ __launch_bounds__(256)
void spdp_policy_kernel(const float* __restrict__ states,
    const float* __restrict__ Wk1, const float* __restrict__ Wq1, const float* __restrict__ Wv1,
    const float* __restrict__ Wk2, const float* __restrict__ Wq2, const float* __restrict__ Wv2,
    const float* __restrict__ Wp1g, const float* __restrict__ Wp2g,
    float* __restrict__ pol, float* __restrict__ w1o, float* __restrict__ w2o)
{
  extern __shared__ __bf16 smem[];
  __bf16* sWq1 = smem;
  __bf16* sWk1 = smem + 16384;
  __bf16* sWv1 = smem + 32768;
  __bf16* sWq2 = smem + 49152;
  __bf16* sWk2 = smem + 65536;
  __bf16* sWv2 = smem + 81920;
  __bf16* sWp1 = smem + 98304;          // 256x128
  __bf16* sWp2 = smem + 131072;         // padded 16x256 (rows 10..15 = 0)

  const int tid = threadIdx.x;
  cp_w(sWq1, Wq1, 16384, tid);
  cp_w(sWk1, Wk1, 16384, tid);
  cp_w(sWv1, Wv1, 16384, tid);
  cp_w(sWq2, Wq2, 16384, tid);
  cp_w(sWk2, Wk2, 16384, tid);
  cp_w(sWv2, Wv2, 16384, tid);
  cp_w(sWp1, Wp1g, 32768, tid);
  for (int i = tid; i < 4096; i += 256) {
    const int r = i >> 8, c = i & 255;
    sWp2[i] = (__bf16)(r < 10 ? Wp2g[r * 256 + c] : 0.0f);
  }
  __syncthreads();

  const int lane = tid & 31, wave = tid >> 5;
  __bf16* scrA = smem + W_ELEMS + wave * SCR_ELEMS;
  __bf16* scrB = scrA + 512;
  __bf16* scrW = scrA + 1024;
  const int half = lane >> 4, col = lane & 15;
  const int stride = gridDim.x * 8;

  for (int tile = blockIdx.x * 8 + wave; tile < TILES; tile += stride) {
    if (tile + stride < TILES)  // speculative prefetch of next tile's rows
      __builtin_prefetch(states + (size_t)(tile + stride) * 16 * 128, 0, 1);

    v16bf xa[4];
#pragma unroll
    for (int kf = 0; kf < 4; ++kf)
      xa[kf] = ld_afrag_global_f32(states, tile * 16, kf * 32, lane);

    v16bf ha[4], na[4];
    attention_layer(xa, sWq1, sWk1, sWv1, w1o, tile, scrA, scrB, scrW, ha, lane);
    attention_layer(ha, sWq2, sWk2, sWv2, w2o, tile, scrA, scrB, scrW, na, lane);

    // hid = leakyrelu(node @ Wp1^T)   [16 x 256] -> A-frags for the 10-way head
    v16bf hidA[8];
#pragma unroll
    for (int p = 0; p < 8; ++p) {
#pragma unroll
      for (int q = 0; q < 2; ++q) {
        const int ht = 2 * p + q;
        v8f hc = {0, 0, 0, 0, 0, 0, 0, 0};
#pragma unroll
        for (int kf = 0; kf < 4; ++kf)
          hc = wmma_bf16(na[kf], ld_bfrag(sWp1, 128, ht * 16, kf * 32, lane), hc);
#pragma unroll
        for (int vi = 0; vi < 8; ++vi)
          hc[vi] = hc[vi] > 0.0f ? hc[vi] : 0.01f * hc[vi];
        store_ctile(scrA, hc, q, lane);
      }
      hidA[p] = ld_afrag(scrA, 32, 0, 0, lane);
    }

    // logits = hid @ Wp2^T (K=256), then softmax over 10 actions
    v8f lg = {0, 0, 0, 0, 0, 0, 0, 0};
#pragma unroll
    for (int p = 0; p < 8; ++p)
      lg = wmma_bf16(hidA[p], ld_bfrag(sWp2, 256, 0, p * 32, lane), lg);

    const bool va = (col < 10);
#pragma unroll
    for (int vi = 0; vi < 8; ++vi) {
      const float v = va ? lg[vi] : -1e30f;
      const float m = rowmax16(v);
      const float e = va ? __expf(v - m) : 0.0f;
      const float sm = rowsum16(e);
      if (va) pol[((size_t)tile * 16 + half * 8 + vi) * 10 + col] = e / sm;
    }
  }
}

extern "C" void kernel_launch(void* const* d_in, const int* in_sizes, int n_in,
                              void* d_out, int out_size, void* d_ws, size_t ws_size,
                              hipStream_t stream) {
  (void)in_sizes; (void)n_in; (void)out_size; (void)d_ws; (void)ws_size;
  // setup_inputs order: states, Wk1, Wq1, Wv1, Wk2, Wq2, Wv2, Wp1, Wp2
  const float* states = (const float*)d_in[0];
  const float* Wk1 = (const float*)d_in[1];
  const float* Wq1 = (const float*)d_in[2];
  const float* Wv1 = (const float*)d_in[3];
  const float* Wk2 = (const float*)d_in[4];
  const float* Wq2 = (const float*)d_in[5];
  const float* Wv2 = (const float*)d_in[6];
  const float* Wp1 = (const float*)d_in[7];
  const float* Wp2 = (const float*)d_in[8];

  float* pol = (float*)d_out;                       // [32768,8,10]
  float* w1  = pol + (size_t)32768 * 8 * 10;        // [32768,8,8,1]
  float* w2  = w1 + (size_t)32768 * 64;             // [32768,8,8,1]

  const size_t smem_bytes = (size_t)SMEM_ELEMS * sizeof(__bf16);  // 294912 B
  (void)hipFuncSetAttribute((const void*)spdp_policy_kernel,
                            hipFuncAttributeMaxDynamicSharedMemorySize,
                            (int)smem_bytes);
  spdp_policy_kernel<<<dim3(1024), dim3(256), smem_bytes, stream>>>(
      states, Wk1, Wq1, Wv1, Wk2, Wq2, Wv2, Wp1, Wp2, pol, w1, w2);
}
